// CosformerAttention_38577396253109
// MI455X (gfx1250) — matrix operand order repaired and checked by reading, not compiled, not run on hardware
//
#include <hip/hip_runtime.h>
#include <hip/hip_bf16.h>
#include <math.h>
#include <stdint.h>

// CDNA5 / gfx1250 cosformer attention, f16-WMMA everywhere.
// Pipeline:
//   0) cvt_kernel      : one-shot f32->f16 conversion of X and *transposed*
//      weights so all GEMM tile staging becomes pure async global->LDS copies.
//   1) qkv_gemm_kernel : Q/K/V = act(X @ W^T + b) scattered into per-head f16
//      buffers with the cosformer sin/cos feature expansion pre-applied.
//   2) attn_kernel     : per-head chunked causal linear attention (C=64), the
//      128x64 f32 KV state lives in WMMA accumulators across chunks.
//   3) out_gemm_kernel : out = Aattn @ Wo^T + bo (f32 output).

typedef __attribute__((ext_vector_type(16))) _Float16 v16h;
typedef __attribute__((ext_vector_type(8)))  _Float16 v8h;
typedef __attribute__((ext_vector_type(8)))  float    v8f;

#define LQ    1024   // sequence length
#define EF    512    // embed dim
#define RROWS 2048   // L * bsz rows in the flat GEMM domain
#define HD    64     // head dim
#define D2    128    // expanded (sin|cos) head dim
#define NHT   16     // bsz * n_head sequences
#define CHK   64     // attention chunk size
#define NCHK  16     // LQ / CHK

__device__ __forceinline__ v8f wmma_f16(v16h a, v16h b, v8f c) {
  // D = A(16x32 f16) * B(32x16 f16) + C(16x16 f32)
  return __builtin_amdgcn_wmma_f32_16x16x32_f16(false, a, false, b, (short)0, c,
                                                false, false);
}

// A-frag: p0 -> 8 contiguous f16 (K = b0..b0+7), p1 -> 8 contiguous (K = b0+16..+23)
// B-frag: see load_bfrag (column n, k = b0..b0+7, b0+16..+23, strided).
__device__ __forceinline__ v16h load_frag(const _Float16* p0, const _Float16* p1) {
  union { v16h v; v8h h[2]; } u;
  u.h[0] = *(const v8h*)p0;
  u.h[1] = *(const v8h*)p1;
  return u.v;
}

__device__ __forceinline__ v16h load_bfrag(const _Float16* p, int ld) {
  v16h v;
#pragma unroll
  for (int s = 0; s < 8; ++s) { v[s] = p[s * ld]; v[8 + s] = p[(16 + s) * ld]; }
  return v;
}

__device__ __forceinline__ v8f zero8() {
  v8f z;
#pragma unroll
  for (int i = 0; i < 8; ++i) z[i] = 0.0f;
  return z;
}

// Async global->LDS 16-byte copy (CDNA5 GLOBAL_LOAD_ASYNC_TO_LDS_B128,
// tracked with ASYNCcnt). Low 32 bits of a generic LDS pointer are the
// wave-relative LDS byte offset (flat aperture truncation rule).
__device__ __forceinline__ void async_copy_b128(void* lds_dst, const void* gsrc) {
  uint32_t ldsoff = (uint32_t)(uintptr_t)lds_dst;
  asm volatile("global_load_async_to_lds_b128 %0, %1, off"
               :
               : "v"(ldsoff), "v"(gsrc)
               : "memory");
}
__device__ __forceinline__ void wait_async() {
  asm volatile("s_wait_asynccnt 0x0" ::: "memory");
}

// ---------------------------------------------------------------------------
// Kernel 0: f32 -> f16 conversion.  z<4: WT[k][j] = W[j][k] (transposed f16
// weights); z>=4: Xh = f16(X) flat quarters.  grid = (1024, 8), block = 256.
// ---------------------------------------------------------------------------
__global__ __launch_bounds__(256) void cvt_kernel(
    const float* __restrict__ Wq, const float* __restrict__ Wk,
    const float* __restrict__ Wv, const float* __restrict__ Wo,
    const float* __restrict__ X,
    _Float16* __restrict__ WTq, _Float16* __restrict__ WTk,
    _Float16* __restrict__ WTv, _Float16* __restrict__ WTo,
    _Float16* __restrict__ Xh) {
  const int z = blockIdx.y;
  const int idx = blockIdx.x * 256 + threadIdx.x;   // 0 .. 262143
  if (z < 4) {
    const float* W = (z == 0) ? Wq : (z == 1) ? Wk : (z == 2) ? Wv : Wo;
    _Float16* WT   = (z == 0) ? WTq : (z == 1) ? WTk : (z == 2) ? WTv : WTo;
    const int j = idx & (EF - 1);
    const int k = idx >> 9;
    WT[(size_t)k * EF + j] = (_Float16)W[(size_t)j * EF + k];
  } else {
    const size_t base = (size_t)(z - 4) * 262144 + idx;
    Xh[base] = (_Float16)X[base];
  }
}

// ---------------------------------------------------------------------------
// Kernel 1: Q/K/V projections. grid = (16, 8, 3), block = 256 (8 waves).
// Workgroup tile: 128(M) x 64(N), K-steps of 32, each wave owns a 16x64 strip.
// All tile staging is async global->LDS b128 (no VGPR round trip).
// ---------------------------------------------------------------------------
__global__ __launch_bounds__(256) void qkv_gemm_kernel(
    const _Float16* __restrict__ Xh,
    const _Float16* __restrict__ WTq, const float* __restrict__ bq,
    const _Float16* __restrict__ WTk, const float* __restrict__ bk,
    const _Float16* __restrict__ WTv, const float* __restrict__ bv,
    _Float16* __restrict__ Qh, _Float16* __restrict__ Kh,
    _Float16* __restrict__ Vh) {
  const int mat = blockIdx.z;
  const _Float16* WT  = (mat == 0) ? WTq : ((mat == 1) ? WTk : WTv);
  const float*   bias = (mat == 0) ? bq : ((mat == 1) ? bk : bv);
  const int R0 = blockIdx.x * 128;
  const int C0 = blockIdx.y * 64;

  __shared__ _Float16 Alds[128 * 40];   // 128 rows x 32 K, stride 40 (pad)
  __shared__ _Float16 BTlds[32 * 72];   // 32 K x 64 N (WT rows), stride 72

  const int t = threadIdx.x;
  const int lane = t & 31, wave = t >> 5;
  const int m16 = lane & 15;
  const int b0  = (lane >> 4) << 3;

  v8f acc[4];
#pragma unroll
  for (int nt = 0; nt < 4; ++nt) acc[nt] = zero8();

  for (int k0 = 0; k0 < EF; k0 += 32) {
    __syncthreads();
    {   // async-stage X tile (2x16B per thread) and WT tile (1x16B per thread)
      const int m = t >> 1, off = (t & 1) * 16;
      const _Float16* src = Xh + (size_t)(R0 + m) * EF + k0 + off;
      async_copy_b128(Alds + m * 40 + off, src);
      async_copy_b128(Alds + m * 40 + off + 8, src + 8);
      const int kr = t >> 3, seg = (t & 7) * 8;
      async_copy_b128(BTlds + kr * 72 + seg,
                      WT + (size_t)(k0 + kr) * EF + C0 + seg);
    }
    wait_async();
    __syncthreads();

    const _Float16* ar = Alds + (wave * 16 + m16) * 40;
    v16h af = load_frag(ar + b0, ar + b0 + 16);
    v16h bf[4];
#pragma unroll
    for (int nt = 0; nt < 4; ++nt) {
      const _Float16* br = BTlds + b0 * 72 + nt * 16 + m16;
      bf[nt] = load_bfrag(br, 72);
    }
#pragma unroll
    for (int nt = 0; nt < 4; ++nt) acc[nt] = wmma_f16(af, bf[nt], acc[nt]);
  }

  // Epilogue: bias (+relu, sin/cos expansion) and head-layout scatter.
#pragma unroll
  for (int i = 0; i < 8; ++i) {
    const int rloc = wave * 16 + ((lane >> 4) << 3) + i;
    const int r = R0 + rloc;
    const int lpos = r >> 1;
    float sv = 0.0f, cv = 0.0f;
    if (mat < 2)
      __sincosf(0.5f * (float)M_PI * (float)(lpos + 1) * (1.0f / (float)LQ),
                &sv, &cv);
#pragma unroll
    for (int nt = 0; nt < 4; ++nt) {
      const int n = C0 + nt * 16 + (lane & 15);
      float val = acc[nt][i] + bias[n];
      const int hh = ((r & 1) << 3) + (n >> 6);
      const int d = n & 63;
      if (mat == 2) {
        Vh[((size_t)hh * LQ + lpos) * HD + d] = (_Float16)val;
      } else {
        val = fmaxf(val, 0.0f);
        _Float16* dst = (mat == 0 ? Qh : Kh) + ((size_t)hh * LQ + lpos) * D2;
        dst[d]      = (_Float16)(val * sv);
        dst[HD + d] = (_Float16)(val * cv);
      }
    }
  }
}

// ---------------------------------------------------------------------------
// Kernel 2: chunked causal linear attention. grid = 16 heads, block = 128.
// Wave w owns output rows [w*16, w*16+16) of each chunk and state rows
// d in [w*32, w*32+32) of the 128x64 f32 KV state (8 WMMA accumulators).
// ---------------------------------------------------------------------------
__global__ __launch_bounds__(128) void attn_kernel(
    const _Float16* __restrict__ Qh, const _Float16* __restrict__ Kh,
    const _Float16* __restrict__ Vh, _Float16* __restrict__ Aat) {
  const int h = blockIdx.x;
  const int t = threadIdx.x, lane = t & 31, wave = t >> 5;
  const int m16 = lane & 15;
  const int b0  = (lane >> 4) << 3;

  __shared__ _Float16 KT[D2 * 72];   // K_ chunk transposed: KT[d][j]
  __shared__ _Float16 Sl[D2 * 72];   // f16 snapshot of state S[d][n]
  __shared__ _Float16 Pl[CHK * 72];  // masked scores P[i][j]
  __shared__ float    zl[D2];        // z[d] = sum_j K_[j][d]
  __shared__ float    dn[CHK];       // per-row denominator

  const _Float16* Qb = Qh + (size_t)h * LQ * D2;
  const _Float16* Kb = Kh + (size_t)h * LQ * D2;
  const _Float16* Vb = Vh + (size_t)h * LQ * HD;

  v8f S[2][4];
#pragma unroll
  for (int mt = 0; mt < 2; ++mt)
#pragma unroll
    for (int nt = 0; nt < 4; ++nt) S[mt][nt] = zero8();
  float zreg = 0.0f;   // z[wave*32 + lane]

  for (int i = t; i < D2 * 72; i += 128) Sl[i] = (_Float16)0.0f;
  if (t < D2) zl[t] = 0.0f;
  __syncthreads();

  for (int c = 0; c < NCHK; ++c) {
    const int l0 = c * CHK;
    __syncthreads();   // prev-chunk consumers of KT done
    if (c + 1 < NCHK) {   // warm next chunk's streams (global_prefetch_b8)
      __builtin_prefetch(Kb + (size_t)(l0 + CHK + t) * D2, 0, 1);
      __builtin_prefetch(Qb + (size_t)(l0 + CHK + t) * D2, 0, 1);
      __builtin_prefetch(Vb + (size_t)(l0 + ((t & 1) ? CHK / 2 : 0) + CHK) * HD, 0, 1);
    }
    {   // stage K_ chunk transposed
      const int j = t >> 1, dd0 = (t & 1) * 64;
      const _Float16* src = Kb + (size_t)(l0 + j) * D2 + dd0;
#pragma unroll
      for (int i = 0; i < 64; ++i) KT[(dd0 + i) * 72 + j] = src[i];
    }
    __syncthreads();

    // denominator, inter part: q_[row] . z_prev  (lane -> row lane&15, d-half lane>>4)
    {
      const int rloc = lane & 15;
      const int d0 = (lane >> 4) * 64;
      const _Float16* q = Qb + (size_t)(l0 + wave * 16 + rloc) * D2 + d0;
      float p = 0.0f;
#pragma unroll
      for (int i = 0; i < 64; ++i) p += (float)q[i] * zl[d0 + i];
      p += __shfl_xor(p, 16, 32);
      if (lane < 16) dn[wave * 16 + rloc] = p;
    }

    // scores: Q_(16x128) @ K_^T(128x64)
    v8f sc[4];
#pragma unroll
    for (int nt = 0; nt < 4; ++nt) sc[nt] = zero8();
#pragma unroll
    for (int ks = 0; ks < 4; ++ks) {
      const int kk = ks * 32;
      const _Float16* qr = Qb + (size_t)(l0 + wave * 16 + m16) * D2 + kk;
      v16h af = load_frag(qr + b0, qr + b0 + 16);
      v16h bf[4];
#pragma unroll
      for (int nt = 0; nt < 4; ++nt) {
        const _Float16* br = KT + (kk + b0) * 72 + nt * 16 + m16;
        bf[nt] = load_bfrag(br, 72);
      }
#pragma unroll
      for (int nt = 0; nt < 4; ++nt) sc[nt] = wmma_f16(af, bf[nt], sc[nt]);
    }
    // causal mask (j <= i), row-sums into denominator, f16 scores -> LDS
    float rs[8];
#pragma unroll
    for (int i = 0; i < 8; ++i) rs[i] = 0.0f;
#pragma unroll
    for (int nt = 0; nt < 4; ++nt) {
#pragma unroll
      for (int i = 0; i < 8; ++i) {
        const int ir = wave * 16 + ((lane >> 4) << 3) + i;
        const int jc = nt * 16 + (lane & 15);
        float v = (jc <= ir) ? sc[nt][i] : 0.0f;
        rs[i] += v;
        Pl[ir * 72 + jc] = (_Float16)v;
      }
    }
#pragma unroll
    for (int i = 0; i < 8; ++i) {
      rs[i] += __shfl_xor(rs[i], 1, 32);
      rs[i] += __shfl_xor(rs[i], 2, 32);
      rs[i] += __shfl_xor(rs[i], 4, 32);
      rs[i] += __shfl_xor(rs[i], 8, 32);
    }
    if ((lane & 15) == 0) {
#pragma unroll
      for (int i = 0; i < 8; ++i)
        dn[wave * 16 + ((lane >> 4) << 3) + i] += rs[i];
    }

    // O = Q_ @ S_prev  +  P @ V
    v8f o[4];
#pragma unroll
    for (int nt = 0; nt < 4; ++nt) o[nt] = zero8();
#pragma unroll
    for (int ks = 0; ks < 4; ++ks) {        // inter, K over d = 0..127
      const int kk = ks * 32;
      const _Float16* qr = Qb + (size_t)(l0 + wave * 16 + m16) * D2 + kk;
      v16h af = load_frag(qr + b0, qr + b0 + 16);
      v16h bf[4];
#pragma unroll
      for (int nt = 0; nt < 4; ++nt) {
        const _Float16* br = Sl + (kk + b0) * 72 + nt * 16 + m16;
        bf[nt] = load_bfrag(br, 72);
      }
#pragma unroll
      for (int nt = 0; nt < 4; ++nt) o[nt] = wmma_f16(af, bf[nt], o[nt]);
    }
#pragma unroll
    for (int ks = 0; ks < 2; ++ks) {        // intra, K over j = 0..63
      const int kk = ks * 32;
      const _Float16* pr = Pl + (wave * 16 + m16) * 72 + kk;
      v16h af = load_frag(pr + b0, pr + b0 + 16);
      v16h bf[4];
#pragma unroll
      for (int nt = 0; nt < 4; ++nt) {
        const _Float16* br = Vb + (size_t)(l0 + kk + b0) * HD + nt * 16 + m16;
        bf[nt] = load_bfrag(br, HD);
      }
#pragma unroll
      for (int nt = 0; nt < 4; ++nt) o[nt] = wmma_f16(af, bf[nt], o[nt]);
    }
    // scale by 1/clip(denom) and scatter to row-major attention output
#pragma unroll
    for (int i = 0; i < 8; ++i) {
      const int ir = wave * 16 + ((lane >> 4) << 3) + i;
      const float inv = 1.0f / fmaxf(dn[ir], 1e-6f);
      const int l = l0 + ir;
      const int r = l * 2 + (h >> 3);
#pragma unroll
      for (int nt = 0; nt < 4; ++nt) {
        const int n = ((h & 7) << 6) + nt * 16 + (lane & 15);
        Aat[(size_t)r * EF + n] = (_Float16)(o[nt][i] * inv);
      }
    }

    __syncthreads();   // everyone done reading Sl / zl of previous state
    // state update: S += K_^T @ V ; z += colsum(K_)
#pragma unroll
    for (int ks = 0; ks < 2; ++ks) {
      const int kk = ks * 32;
      v16h bf[4];
#pragma unroll
      for (int nt = 0; nt < 4; ++nt) {
        const _Float16* br = Vb + (size_t)(l0 + kk + b0) * HD + nt * 16 + m16;
        bf[nt] = load_bfrag(br, HD);
      }
#pragma unroll
      for (int mt = 0; mt < 2; ++mt) {
        const _Float16* ar = KT + (wave * 32 + mt * 16 + m16) * 72 + kk;
        v16h af = load_frag(ar + b0, ar + b0 + 16);
#pragma unroll
        for (int nt = 0; nt < 4; ++nt)
          S[mt][nt] = wmma_f16(af, bf[nt], S[mt][nt]);
      }
    }
    {
      const v8h* kr = (const v8h*)(KT + (wave * 32 + lane) * 72);
      float s = 0.0f;
#pragma unroll
      for (int q = 0; q < 8; ++q) {
        v8h x = kr[q];
#pragma unroll
        for (int j = 0; j < 8; ++j) s += (float)x[j];
      }
      zreg += s;
      zl[wave * 32 + lane] = zreg;
    }
#pragma unroll
    for (int mt = 0; mt < 2; ++mt)
#pragma unroll
      for (int nt = 0; nt < 4; ++nt)
#pragma unroll
        for (int i = 0; i < 8; ++i) {
          const int dloc = wave * 32 + mt * 16 + ((lane >> 4) << 3) + i;
          const int n = nt * 16 + (lane & 15);
          Sl[dloc * 72 + n] = (_Float16)S[mt][nt][i];
        }
  }
}

// ---------------------------------------------------------------------------
// Kernel 3: out = Aattn(f16, 2048x512) @ Wo^T + bo, f32 output.
// Same tiling as kernel 1, async staging from f16 Aattn and transposed WTo.
// ---------------------------------------------------------------------------
__global__ __launch_bounds__(256) void out_gemm_kernel(
    const _Float16* __restrict__ A, const _Float16* __restrict__ WTo,
    const float* __restrict__ bo, float* __restrict__ out) {
  const int R0 = blockIdx.x * 128;
  const int C0 = blockIdx.y * 64;
  __shared__ _Float16 Alds[128 * 40];
  __shared__ _Float16 BTlds[32 * 72];

  const int t = threadIdx.x;
  const int lane = t & 31, wave = t >> 5;
  const int m16 = lane & 15;
  const int b0  = (lane >> 4) << 3;

  v8f acc[4];
#pragma unroll
  for (int nt = 0; nt < 4; ++nt) acc[nt] = zero8();

  for (int k0 = 0; k0 < EF; k0 += 32) {
    __syncthreads();
    {
      const int m = t >> 1, off = (t & 1) * 16;
      const _Float16* src = A + (size_t)(R0 + m) * EF + k0 + off;
      async_copy_b128(Alds + m * 40 + off, src);
      async_copy_b128(Alds + m * 40 + off + 8, src + 8);
      const int kr = t >> 3, seg = (t & 7) * 8;
      async_copy_b128(BTlds + kr * 72 + seg,
                      WTo + (size_t)(k0 + kr) * EF + C0 + seg);
    }
    wait_async();
    __syncthreads();

    const _Float16* ar = Alds + (wave * 16 + m16) * 40;
    v16h af = load_frag(ar + b0, ar + b0 + 16);
    v16h bf[4];
#pragma unroll
    for (int nt = 0; nt < 4; ++nt) {
      const _Float16* br = BTlds + b0 * 72 + nt * 16 + m16;
      bf[nt] = load_bfrag(br, 72);
    }
#pragma unroll
    for (int nt = 0; nt < 4; ++nt) acc[nt] = wmma_f16(af, bf[nt], acc[nt]);
  }

#pragma unroll
  for (int i = 0; i < 8; ++i) {
    const int r = R0 + wave * 16 + ((lane >> 4) << 3) + i;
#pragma unroll
    for (int nt = 0; nt < 4; ++nt) {
      const int n = C0 + nt * 16 + (lane & 15);
      out[(size_t)r * EF + n] = acc[nt][i] + bo[n];
    }
  }
}

extern "C" void kernel_launch(void* const* d_in, const int* in_sizes, int n_in,
                              void* d_out, int out_size, void* d_ws,
                              size_t ws_size, hipStream_t stream) {
  const float* X  = (const float*)d_in[0];
  const float* Wq = (const float*)d_in[1];
  const float* bq = (const float*)d_in[2];
  const float* Wk = (const float*)d_in[3];
  const float* bk = (const float*)d_in[4];
  const float* Wv = (const float*)d_in[5];
  const float* bv = (const float*)d_in[6];
  const float* Wo = (const float*)d_in[7];
  const float* bo = (const float*)d_in[8];
  float* out = (float*)d_out;

  char* ws = (char*)d_ws;
  const size_t MiB = 1024 * 1024;
  _Float16* Qh  = (_Float16*)(ws);              // 16*1024*128 f16 = 4 MiB
  _Float16* Kh  = (_Float16*)(ws + 4 * MiB);    // 4 MiB
  _Float16* Vh  = (_Float16*)(ws + 8 * MiB);    // 2 MiB
  _Float16* Aat = (_Float16*)(ws + 10 * MiB);   // 2 MiB
  _Float16* WTq = (_Float16*)(ws + 12 * MiB);   // 512 KiB each (transposed f16)
  _Float16* WTk = (_Float16*)(ws + 12 * MiB + 512 * 1024);
  _Float16* WTv = (_Float16*)(ws + 13 * MiB);
  _Float16* WTo = (_Float16*)(ws + 13 * MiB + 512 * 1024);
  _Float16* Xh  = (_Float16*)(ws + 14 * MiB);   // 2 MiB

  dim3 g0(1024, 8);
  cvt_kernel<<<g0, 256, 0, stream>>>(Wq, Wk, Wv, Wo, X, WTq, WTk, WTv, WTo, Xh);
  dim3 g1(RROWS / 128, EF / 64, 3);
  qkv_gemm_kernel<<<g1, 256, 0, stream>>>(Xh, WTq, bq, WTk, bk, WTv, bv,
                                          Qh, Kh, Vh);
  attn_kernel<<<NHT, 128, 0, stream>>>(Qh, Kh, Vh, Aat);
  dim3 g3(RROWS / 128, EF / 64);
  out_gemm_kernel<<<g3, 256, 0, stream>>>(Aat, WTo, bo, out);

  (void)in_sizes; (void)n_in; (void)out_size; (void)ws_size;
}
